// TransformerMambaBlock_19387482374995
// MI455X (gfx1250) — matrix operand order repaired and checked by reading, not compiled, not run on hardware
//
#include <hip/hip_runtime.h>
#include <math.h>

#define BB 2
#define CC 128
#define HS 96
#define NN (HS*HS)          // 9216
#define RR (BB*NN)          // 18432
#define MLPD 512
#define DH 64
#define DS 8
#define NCH 72
#define CLEN 128            // NN / NCH

typedef __bf16 bf16;
typedef __attribute__((ext_vector_type(16))) __bf16 v16bf;
typedef __attribute__((ext_vector_type(8)))  __bf16 v8bf;
typedef __attribute__((ext_vector_type(8)))  float  v8f;

union frag_u { v16bf v; v8bf h[2]; };

__device__ __forceinline__ bf16 f2b(float f){
  union { float f; unsigned u; } a; a.f = f;
  unsigned r = a.u + 0x7FFFu + ((a.u >> 16) & 1u);
  union { unsigned short s; bf16 b; } o; o.s = (unsigned short)(r >> 16);
  return o.b;
}
__device__ __forceinline__ float siluf(float x){ return x / (1.f + expf(-x)); }

// ---------------------------------------------------------------- WMMA GEMM
// C[M,N] = A[M,K](bf16,row) * W[N,K](bf16,row)^T (+bias[N]) (+res[M,N])
__device__ __forceinline__ v16bf load_fragA(const bf16* __restrict__ p, int row0, int K, int k0){
  int lane = threadIdx.x & 31;
  const bf16* q = p + (size_t)(row0 + (lane & 15)) * K + k0 + ((lane >> 4) << 3);
  frag_u f;
  f.h[0] = *reinterpret_cast<const v8bf*>(q);
  f.h[1] = *reinterpret_cast<const v8bf*>(q + 16);
  return f.v;
}
__device__ __forceinline__ v16bf load_fragB(const bf16* __restrict__ p, int col0, int K, int k0){
  int lane = threadIdx.x & 31;
  const bf16* q = p + (size_t)(col0 + (lane & 15)) * K + k0 + ((lane >> 4) << 3);
  frag_u f;
  f.h[0] = *reinterpret_cast<const v8bf*>(q);
  f.h[1] = *reinterpret_cast<const v8bf*>(q + 16);
  return f.v;
}

__global__ __launch_bounds__(256) void gemm_bf16_kernel(
    const bf16* __restrict__ A, const bf16* __restrict__ Bw,
    const float* __restrict__ bias, const float* __restrict__ res,
    float* __restrict__ Co, int M, int N, int K)
{
  int wave = threadIdx.x >> 5;
  int wm = blockIdx.y * 128 + (wave >> 1) * 32;
  int wn = blockIdx.x * 64  + (wave & 1) * 32;
  v8f a00 = {0,0,0,0,0,0,0,0}, a01 = a00, a10 = a00, a11 = a00;
  for (int k0 = 0; k0 < K; k0 += 32){
    v16bf fa0 = load_fragA(A,  wm,      K, k0);
    v16bf fa1 = load_fragA(A,  wm + 16, K, k0);
    v16bf fb0 = load_fragB(Bw, wn,      K, k0);
    v16bf fb1 = load_fragB(Bw, wn + 16, K, k0);
    a00 = __builtin_amdgcn_wmma_f32_16x16x32_bf16(false, fa0, false, fb0, (short)0, a00, false, false);
    a01 = __builtin_amdgcn_wmma_f32_16x16x32_bf16(false, fa0, false, fb1, (short)0, a01, false, false);
    a10 = __builtin_amdgcn_wmma_f32_16x16x32_bf16(false, fa1, false, fb0, (short)0, a10, false, false);
    a11 = __builtin_amdgcn_wmma_f32_16x16x32_bf16(false, fa1, false, fb1, (short)0, a11, false, false);
  }
  int lane = threadIdx.x & 31;
  int nIn = lane & 15, mh = (lane >> 4) * 8;
  auto store_tile = [&](v8f acc, int mBase, int nBase){
    #pragma unroll
    for (int r2 = 0; r2 < 8; r2++){
      int m = mBase + mh + r2, n = nBase + nIn;
      float v = acc[r2];
      if (bias) v += bias[n];
      size_t o = (size_t)m * N + n;
      if (res)  v += res[o];
      Co[o] = v;
    }
  };
  store_tile(a00, wm, wn);      store_tile(a01, wm, wn + 16);
  store_tile(a10, wm + 16, wn); store_tile(a11, wm + 16, wn + 16);
}

// ---------------------------------------------------------------- LayerNorm (C=128)
__global__ __launch_bounds__(128) void ln_kernel(const float* __restrict__ x,
    const float* __restrict__ g, const float* __restrict__ b,
    float* __restrict__ outf, bf16* __restrict__ outb)
{
  int row = blockIdx.x, t = threadIdx.x;
  float v = x[(size_t)row * CC + t];
  __shared__ float red[128];
  red[t] = v; __syncthreads();
  for (int st = 64; st > 0; st >>= 1){ if (t < st) red[t] += red[t + st]; __syncthreads(); }
  float mu = red[0] * (1.f / CC);
  __syncthreads();
  float d = v - mu;
  red[t] = d * d; __syncthreads();
  for (int st = 64; st > 0; st >>= 1){ if (t < st) red[t] += red[t + st]; __syncthreads(); }
  float var = red[0] * (1.f / CC);
  float y = d * rsqrtf(var + 1e-5f) * g[t] + b[t];
  if (outf) outf[(size_t)row * CC + t] = y;
  if (outb) outb[(size_t)row * CC + t] = f2b(y);
}

// ---------------------------------------------------------------- layout helpers
__global__ void to_blc_kernel(const float* __restrict__ src, float* __restrict__ dst){
  size_t i = (size_t)blockIdx.x * 256 + threadIdx.x;
  if (i >= (size_t)BB * NN * CC) return;
  int ch = (int)(i % CC);
  size_t j = i / CC;
  int n = (int)(j % NN), b = (int)(j / NN);
  dst[i] = src[((size_t)b * CC + ch) * NN + n];
}
__global__ void final_kernel(const float* __restrict__ xtr, const float* __restrict__ nm, float* __restrict__ out){
  size_t i = (size_t)blockIdx.x * 256 + threadIdx.x;
  if (i >= (size_t)BB * CC * NN) return;
  int n = (int)(i % NN);
  size_t j = i / NN;
  int ch = (int)(j % CC), b = (int)(j / CC);
  size_t s = ((size_t)b * NN + n) * CC + ch;
  out[i] = xtr[s] + nm[s];
}
__global__ void add_kernel(const float* __restrict__ a, const float* __restrict__ b, float* __restrict__ o, int n){
  int i = blockIdx.x * 256 + threadIdx.x;
  if (i < n) o[i] = a[i] + b[i];
}
__global__ void cvt_bf16_kernel(const float* __restrict__ a, bf16* __restrict__ o, int n){
  int i = blockIdx.x * 256 + threadIdx.x;
  if (i < n) o[i] = f2b(a[i]);
}

// ---------------------------------------------------------------- dwconv2d 3x3 (SAME), [B,N,3C] -> [B,3C,N]
__global__ __launch_bounds__(256) void dwconv_kernel(const float* __restrict__ qkv,
    const float* __restrict__ w, const float* __restrict__ bias, float* __restrict__ out)
{
  const int OC = 3 * CC;
  int bo = blockIdx.y;
  int b = bo / OC, o = bo % OC;
  int n0 = blockIdx.x * 256;
  __shared__ float sm[256 + 194];
  for (int idx = threadIdx.x; idx < 450; idx += 256){
    int n = n0 - 97 + idx;
    sm[idx] = (n >= 0 && n < NN) ? qkv[((size_t)b * NN + n) * OC + o] : 0.f;
  }
  __syncthreads();
  int n = n0 + threadIdx.x;
  int h = n / HS, wc = n % HS;
  float a = bias[o];
  #pragma unroll
  for (int dy = -1; dy <= 1; dy++)
    #pragma unroll
    for (int dx = -1; dx <= 1; dx++){
      int hh = h + dy, ww = wc + dx;
      if (hh >= 0 && hh < HS && ww >= 0 && ww < HS)
        a += sm[threadIdx.x + 97 + dy * HS + dx] * w[o * 9 + (dy + 1) * 3 + (dx + 1)];
    }
  out[((size_t)b * OC + o) * NN + n] = a;
}

// ---------------------------------------------------------------- q,k row L2 normalize in-place
__global__ __launch_bounds__(256) void rownorm_kernel(float* __restrict__ qdw){
  int row = blockIdx.x;                     // 0..B*2C-1
  int b = row / (2 * CC), rr = row % (2 * CC);
  float* p = qdw + ((size_t)b * 3 * CC + rr) * (size_t)NN;
  __shared__ float red[256];
  float s = 0.f;
  for (int i = threadIdx.x; i < NN; i += 256){ float v = p[i]; s += v * v; }
  red[threadIdx.x] = s; __syncthreads();
  for (int st = 128; st > 0; st >>= 1){ if (threadIdx.x < st) red[threadIdx.x] += red[threadIdx.x + st]; __syncthreads(); }
  float inv = 1.f / fmaxf(sqrtf(red[0]), 1e-12f);
  for (int i = threadIdx.x; i < NN; i += 256) p[i] *= inv;
}

// ---------------------------------------------------------------- fused channel attention per (b,head)
__global__ __launch_bounds__(256) void attn_kernel(const float* __restrict__ qdw,
    const float* __restrict__ temp, bf16* __restrict__ outb)
{
  int b = blockIdx.x >> 2, hh = blockIdx.x & 3;
  const float* q = qdw + ((size_t)b * 3 * CC + hh * 32) * (size_t)NN;
  const float* k = q + (size_t)CC * NN;
  const float* v = k + (size_t)CC * NN;
  __shared__ float qt[32][33];
  __shared__ float kt[32][33];
  __shared__ float Sm[32][33];
  int tid = threadIdx.x;
  int c = tid >> 3, g8 = tid & 7;
  float acc0 = 0, acc1 = 0, acc2 = 0, acc3 = 0;
  for (int n0 = 0; n0 < NN; n0 += 32){
    #pragma unroll
    for (int j = 0; j < 4; j++){
      int idx = tid + j * 256;
      int r = idx >> 5, col = idx & 31;
      qt[r][col] = q[(size_t)r * NN + n0 + col];
      kt[r][col] = k[(size_t)r * NN + n0 + col];
    }
    __syncthreads();
    #pragma unroll 8
    for (int nn = 0; nn < 32; nn++){
      float qv = qt[c][nn];
      acc0 += qv * kt[g8 * 4 + 0][nn];
      acc1 += qv * kt[g8 * 4 + 1][nn];
      acc2 += qv * kt[g8 * 4 + 2][nn];
      acc3 += qv * kt[g8 * 4 + 3][nn];
    }
    __syncthreads();
  }
  float tv = temp[hh];
  Sm[c][g8 * 4 + 0] = acc0 * tv; Sm[c][g8 * 4 + 1] = acc1 * tv;
  Sm[c][g8 * 4 + 2] = acc2 * tv; Sm[c][g8 * 4 + 3] = acc3 * tv;
  __syncthreads();
  if (tid < 32){
    float mx = -1e30f;
    for (int j = 0; j < 32; j++) mx = fmaxf(mx, Sm[tid][j]);
    float s = 0.f;
    for (int j = 0; j < 32; j++){ float e = expf(Sm[tid][j] - mx); Sm[tid][j] = e; s += e; }
    float inv = 1.f / s;
    for (int j = 0; j < 32; j++) Sm[tid][j] *= inv;
  }
  __syncthreads();
  for (int n0 = 0; n0 < NN; n0 += 32){
    #pragma unroll
    for (int j = 0; j < 4; j++){
      int idx = tid + j * 256;
      int r = idx >> 5, col = idx & 31;
      qt[r][col] = v[(size_t)r * NN + n0 + col];
    }
    __syncthreads();
    float o0 = 0, o1 = 0, o2 = 0, o3 = 0;
    #pragma unroll 8
    for (int d = 0; d < 32; d++){
      float s = Sm[c][d];
      o0 += s * qt[d][g8 * 4 + 0];
      o1 += s * qt[d][g8 * 4 + 1];
      o2 += s * qt[d][g8 * 4 + 2];
      o3 += s * qt[d][g8 * 4 + 3];
    }
    size_t base = ((size_t)b * NN + n0) * CC + hh * 32 + c;
    outb[base + (size_t)(g8 * 4 + 0) * CC] = f2b(o0);
    outb[base + (size_t)(g8 * 4 + 1) * CC] = f2b(o1);
    outb[base + (size_t)(g8 * 4 + 2) * CC] = f2b(o2);
    outb[base + (size_t)(g8 * 4 + 3) * CC] = f2b(o3);
    __syncthreads();
  }
}

// ---------------------------------------------------------------- KAN: silu + cubic B-spline bases -> bf16 concat [rows, 9F]
__global__ void splinecat_kernel(const float* __restrict__ z, bf16* __restrict__ acat, int rows, int F){
  size_t i = (size_t)blockIdx.x * blockDim.x + threadIdx.x;
  if (i >= (size_t)rows * F) return;
  int fi = (int)(i % F);
  size_t r = i / F;
  float x = z[i];
  bf16* row = acat + r * (size_t)(9 * F);
  row[fi] = f2b(siluf(x));
  const float hg = 0.4f;
  float bb[11];
  #pragma unroll
  for (int j = 0; j < 11; j++){
    float gj = -1.f + (j - 3) * hg;
    bb[j] = (x >= gj && x < gj + hg) ? 1.f : 0.f;
  }
  #pragma unroll
  for (int kk = 1; kk <= 3; kk++){
    float inv = 1.f / (kk * hg);
    for (int j = 0; j <= 10 - kk; j++){
      float gj   = -1.f + (j - 3) * hg;
      float gjk1 = -1.f + (j + kk - 2) * hg;
      bb[j] = (x - gj) * inv * bb[j] + (gjk1 - x) * inv * bb[j + 1];
    }
  }
  union { v8bf v; bf16 e[8]; } st;
  #pragma unroll
  for (int cs = 0; cs < 8; cs++) st.e[cs] = f2b(bb[cs]);
  *reinterpret_cast<v8bf*>(row + F + (size_t)fi * 8) = st.v;
}

// weight concat: [bw | sw*sc] -> bf16 [O, 9F]
__global__ void wcat_kernel(const float* __restrict__ bw, const float* __restrict__ sw,
    const float* __restrict__ sc, bf16* __restrict__ wc, int O, int F)
{
  int i = blockIdx.x * blockDim.x + threadIdx.x;
  if (i >= O * F) return;
  int fi = i % F, o = i / F;
  bf16* row = wc + (size_t)o * (9 * F);
  row[fi] = f2b(bw[i]);
  float s = sc[i];
  const float* sp = sw + (size_t)i * 8;
  union { v8bf v; bf16 e[8]; } st;
  #pragma unroll
  for (int cb = 0; cb < 8; cb++) st.e[cb] = f2b(sp[cb] * s);
  *reinterpret_cast<v8bf*>(row + F + (size_t)fi * 8) = st.v;
}

// ---------------------------------------------------------------- Mamba pieces
__global__ void conv1d_kernel(const float* __restrict__ xz, const float* __restrict__ wx,
    const float* __restrict__ wz, float* __restrict__ xl, float* __restrict__ zl)
{
  size_t i = (size_t)blockIdx.x * 256 + threadIdx.x;
  if (i >= (size_t)BB * NN * DH) return;
  int d = (int)(i & 63);
  size_t j = i >> 6;
  int l = (int)(j % NN), b = (int)(j / NN);
  const float* base = xz + (size_t)b * NN * CC;
  float x0 = (l > 0)      ? base[(size_t)(l - 1) * CC + d] : 0.f;
  float x1 =                base[(size_t)l * CC + d];
  float x2 = (l < NN - 1) ? base[(size_t)(l + 1) * CC + d] : 0.f;
  xl[i] = siluf(x0 * wx[d * 3 + 0] + x1 * wx[d * 3 + 1] + x2 * wx[d * 3 + 2]);
  float z0 = (l > 0)      ? base[(size_t)(l - 1) * CC + DH + d] : 0.f;
  float z1 =                base[(size_t)l * CC + DH + d];
  float z2 = (l < NN - 1) ? base[(size_t)(l + 1) * CC + DH + d] : 0.f;
  zl[i] = siluf(z0 * wz[d * 3 + 0] + z1 * wz[d * 3 + 1] + z2 * wz[d * 3 + 2]);
}
__global__ void xproj_kernel(const float* __restrict__ xl, const float* __restrict__ xpw, float* __restrict__ xdbl){
  size_t i = (size_t)blockIdx.x * 256 + threadIdx.x;
  if (i >= (size_t)RR * 24) return;
  int jj = (int)(i % 24);
  size_t r = i / 24;
  const float* xr = xl + r * DH;
  const float* wr = xpw + (size_t)jj * DH;
  float s = 0.f;
  #pragma unroll 8
  for (int k = 0; k < DH; k++) s += xr[k] * wr[k];
  xdbl[i] = s;
}
__global__ void dt_kernel(const float* __restrict__ xdbl, const float* __restrict__ dtw,
    const float* __restrict__ dtb, float* __restrict__ delta)
{
  size_t i = (size_t)blockIdx.x * 256 + threadIdx.x;
  if (i >= (size_t)RR * DH) return;
  int d = (int)(i & 63);
  size_t r = i >> 6;
  const float* xr = xdbl + r * 24;
  float s = dtb[d];
  #pragma unroll
  for (int k = 0; k < 8; k++) s += xr[k] * dtw[d * 8 + k];
  delta[i] = (s > 20.f) ? s : log1pf(expf(s));
}
__global__ void aneg_kernel(const float* __restrict__ alog, float* __restrict__ an){
  int i = blockIdx.x * 256 + threadIdx.x;
  if (i < DH * DS) an[i] = -expf(alog[i]);
}
// chunked associative scan: pass1 per-chunk (prodA, h_local)
__global__ __launch_bounds__(256) void scan1_kernel(const float* __restrict__ delta,
    const float* __restrict__ xdbl, const float* __restrict__ xl, const float* __restrict__ An,
    float* __restrict__ P, float* __restrict__ Hc)
{
  int idx = blockIdx.x * 256 + threadIdx.x;   // [b,d,n,ch]
  int ch = idx % NCH;
  int rest = idx / NCH;
  int n = rest & 7, d = (rest >> 3) & 63, b = rest >> 9;
  float a_dn = An[d * DS + n];
  float Pv = 1.f, hv = 0.f;
  int l0 = ch * CLEN;
  for (int l = l0; l < l0 + CLEN; l++){
    size_t t = (size_t)b * NN + l;
    float dl = delta[t * DH + d];
    float Bv = xdbl[t * 24 + 8 + n];
    float xv = xl[t * DH + d];
    float da = expf(dl * a_dn);
    Pv *= da;
    hv = da * hv + dl * Bv * xv;
  }
  P[idx] = Pv; Hc[idx] = hv;
}
__global__ void scan2_kernel(const float* __restrict__ P, const float* __restrict__ Hc, float* __restrict__ Hi){
  int idx = blockIdx.x * 256 + threadIdx.x;   // < B*DH*DS
  size_t base = (size_t)idx * NCH;
  float hv = 0.f;
  for (int ch = 0; ch < NCH; ch++){
    Hi[base + ch] = hv;
    hv = P[base + ch] * hv + Hc[base + ch];
  }
}
__global__ __launch_bounds__(256) void scan3_kernel(const float* __restrict__ delta,
    const float* __restrict__ xdbl, const float* __restrict__ xl, const float* __restrict__ An,
    const float* __restrict__ Hi, float* __restrict__ y)
{
  int idx = blockIdx.x * 256 + threadIdx.x;   // n fastest for 8-lane reduction
  int n = idx & 7;
  int rest = idx >> 3;
  int ch = rest % NCH;
  int rest2 = rest / NCH;
  int d = rest2 & 63, b = rest2 >> 6;
  float a_dn = An[d * DS + n];
  float hv = Hi[((size_t)(b * DH + d) * DS + n) * NCH + ch];
  int l0 = ch * CLEN;
  for (int l = l0; l < l0 + CLEN; l++){
    size_t t = (size_t)b * NN + l;
    float dl = delta[t * DH + d];
    float Bv = xdbl[t * 24 + 8 + n];
    float Cv = xdbl[t * 24 + 16 + n];
    float xv = xl[t * DH + d];
    float da = expf(dl * a_dn);
    hv = da * hv + dl * Bv * xv;
    float part = hv * Cv;
    part += __shfl_xor(part, 1, 32);
    part += __shfl_xor(part, 2, 32);
    part += __shfl_xor(part, 4, 32);
    if (n == 0) y[t * DH + d] = part;
  }
}
__global__ void ycat_kernel(const float* __restrict__ y, const float* __restrict__ xl,
    const float* __restrict__ zl, const float* __restrict__ Dp, bf16* __restrict__ yc)
{
  size_t i = (size_t)blockIdx.x * 256 + threadIdx.x;
  if (i >= (size_t)RR * DH) return;
  int d = (int)(i & 63);
  size_t r = i >> 6;
  yc[r * CC + d]      = f2b(y[i] + xl[i] * Dp[d]);
  yc[r * CC + DH + d] = f2b(zl[i]);
}

// ---------------------------------------------------------------- host side
static inline unsigned cdiv(size_t a, unsigned b){ return (unsigned)((a + b - 1) / b); }

static void launch_gemm(const bf16* A, const bf16* Bw, const float* bias, const float* res,
                        float* C, int M, int N, int K, hipStream_t s)
{
  dim3 g(N / 64, M / 128);
  gemm_bf16_kernel<<<g, 256, 0, s>>>(A, Bw, bias, res, C, M, N, K);
}

static void fkan_block(const float* xin, const float* ng, const float* nb,
                       const float* bw0, const float* sw0, const float* sc0,
                       const float* bw1, const float* sw1, const float* sc1,
                       const float* resid, float* out,
                       float* lnf2, bf16* acat, float* hbuf,
                       bf16* wcat0, bf16* wcat1, hipStream_t s)
{
  ln_kernel<<<RR, 128, 0, s>>>(xin, ng, nb, lnf2, nullptr);
  wcat_kernel<<<cdiv((size_t)MLPD * CC, 256), 256, 0, s>>>(bw0, sw0, sc0, wcat0, MLPD, CC);
  wcat_kernel<<<cdiv((size_t)CC * MLPD, 256), 256, 0, s>>>(bw1, sw1, sc1, wcat1, CC, MLPD);
  splinecat_kernel<<<cdiv((size_t)RR * CC, 256), 256, 0, s>>>(lnf2, acat, RR, CC);
  launch_gemm(acat, wcat0, nullptr, nullptr, hbuf, RR, MLPD, 9 * CC, s);
  for (int c = 0; c < 4; c++){
    size_t r0 = (size_t)c * 4608;
    splinecat_kernel<<<cdiv((size_t)4608 * MLPD, 256), 256, 0, s>>>(hbuf + r0 * MLPD, acat, 4608, MLPD);
    launch_gemm(acat, wcat1, nullptr, resid + r0 * CC, out + r0 * CC, 4608, CC, 9 * MLPD, s);
  }
}

extern "C" void kernel_launch(void* const* d_in, const int* in_sizes, int n_in,
                              void* d_out, int out_size, void* d_ws, size_t ws_size,
                              hipStream_t stream)
{
  (void)in_sizes; (void)n_in; (void)out_size; (void)ws_size;
  const float* x5d      = (const float*)d_in[0];
  const float* ln1_g    = (const float*)d_in[1];
  const float* ln1_b    = (const float*)d_in[2];
  const float* ln2_g    = (const float*)d_in[3];
  const float* ln2_b    = (const float*)d_in[4];
  const float* ln3_g    = (const float*)d_in[5];
  const float* ln3_b    = (const float*)d_in[6];
  const float* ln4_g    = (const float*)d_in[7];
  const float* ln4_b    = (const float*)d_in[8];
  const float* attn_temp  = (const float*)d_in[9];
  const float* attn_qkv_w = (const float*)d_in[10];
  const float* attn_qkv_b = (const float*)d_in[11];
  const float* attn_dw_w  = (const float*)d_in[12];
  const float* attn_dw_b  = (const float*)d_in[13];
  const float* attn_po_w  = (const float*)d_in[14];
  const float* attn_po_b  = (const float*)d_in[15];
  const float* f1_ng = (const float*)d_in[16];
  const float* f1_nb = (const float*)d_in[17];
  const float* f1_bw0 = (const float*)d_in[18];
  const float* f1_sw0 = (const float*)d_in[19];
  const float* f1_sc0 = (const float*)d_in[20];
  const float* f1_bw1 = (const float*)d_in[21];
  const float* f1_sw1 = (const float*)d_in[22];
  const float* f1_sc1 = (const float*)d_in[23];
  const float* f2_ng = (const float*)d_in[24];
  const float* f2_nb = (const float*)d_in[25];
  const float* f2_bw0 = (const float*)d_in[26];
  const float* f2_sw0 = (const float*)d_in[27];
  const float* f2_sc0 = (const float*)d_in[28];
  const float* f2_bw1 = (const float*)d_in[29];
  const float* f2_sw1 = (const float*)d_in[30];
  const float* f2_sc1 = (const float*)d_in[31];
  const float* in_proj_w  = (const float*)d_in[32];
  const float* x_proj_w   = (const float*)d_in[33];
  const float* dt_proj_w  = (const float*)d_in[34];
  const float* dt_proj_b  = (const float*)d_in[35];
  const float* A_log      = (const float*)d_in[36];
  const float* D_param    = (const float*)d_in[37];
  const float* out_proj_w = (const float*)d_in[38];
  const float* conv_x_w   = (const float*)d_in[39];
  const float* conv_z_w   = (const float*)d_in[40];

  char* wp = (char*)d_ws;
  size_t off = 0;
  auto alloc = [&](size_t bytes) -> void* {
    off = (off + 255) & ~(size_t)255;
    void* p = wp + off;
    off += bytes;
    return p;
  };
  const size_t FRC = (size_t)RR * CC * 4;
  float* xb    = (float*)alloc(FRC);
  float* tb    = (float*)alloc(FRC);
  float* ub    = (float*)alloc(FRC);
  float* xtrb  = (float*)alloc(FRC);
  float* mb    = (float*)alloc(FRC);
  float* nmb   = (float*)alloc(FRC);
  float* lnf   = (float*)alloc(FRC);
  float* lnf2  = (float*)alloc(FRC);
  float* xzb   = (float*)alloc(FRC);
  float* qkvb  = (float*)alloc((size_t)RR * 384 * 4);
  float* qdwb  = (float*)alloc((size_t)RR * 384 * 4);
  float* hbuf  = (float*)alloc((size_t)RR * MLPD * 4);
  float* xlb   = (float*)alloc((size_t)RR * DH * 4);
  float* zlb   = (float*)alloc((size_t)RR * DH * 4);
  float* xdblb = (float*)alloc((size_t)RR * 24 * 4);
  float* deltab= (float*)alloc((size_t)RR * DH * 4);
  float* yb    = (float*)alloc((size_t)RR * DH * 4);
  bf16*  lnbf  = (bf16*) alloc((size_t)RR * CC * 2);
  bf16*  attnbf= (bf16*) alloc((size_t)RR * CC * 2);
  bf16*  ycatb = (bf16*) alloc((size_t)RR * CC * 2);
  bf16*  acatb = (bf16*) alloc((size_t)RR * 9 * CC * 2);    // reused as 4608 x 4608
  bf16*  wcat0b= (bf16*) alloc((size_t)MLPD * 9 * CC * 2);
  bf16*  wcat1b= (bf16*) alloc((size_t)CC * 9 * MLPD * 2);
  bf16*  wqkvb = (bf16*) alloc((size_t)384 * CC * 2);
  bf16*  wpob  = (bf16*) alloc((size_t)CC * CC * 2);
  bf16*  winpb = (bf16*) alloc((size_t)CC * CC * 2);
  bf16*  woutpb= (bf16*) alloc((size_t)CC * CC * 2);
  float* anegb = (float*)alloc((size_t)DH * DS * 4);
  float* scanP = (float*)alloc((size_t)BB * DH * DS * NCH * 4);
  float* scanHc= (float*)alloc((size_t)BB * DH * DS * NCH * 4);
  float* scanHi= (float*)alloc((size_t)BB * DH * DS * NCH * 4);

  const size_t EL = (size_t)RR * CC;

  // x [B,N,C]
  to_blc_kernel<<<cdiv(EL, 256), 256, 0, stream>>>(x5d, xb);

  // ---- attention branch: t = x + attn(LN1(x))
  ln_kernel<<<RR, 128, 0, stream>>>(xb, ln1_g, ln1_b, nullptr, lnbf);
  cvt_bf16_kernel<<<cdiv(384 * CC, 256), 256, 0, stream>>>(attn_qkv_w, wqkvb, 384 * CC);
  cvt_bf16_kernel<<<cdiv(CC * CC, 256), 256, 0, stream>>>(attn_po_w, wpob, CC * CC);
  launch_gemm(lnbf, wqkvb, attn_qkv_b, nullptr, qkvb, RR, 384, CC, stream);
  dwconv_kernel<<<dim3(NN / 256, BB * 384), 256, 0, stream>>>(qkvb, attn_dw_w, attn_dw_b, qdwb);
  rownorm_kernel<<<BB * 2 * CC, 256, 0, stream>>>(qdwb);
  attn_kernel<<<BB * 4, 256, 0, stream>>>(qdwb, attn_temp, attnbf);
  launch_gemm(attnbf, wpob, attn_po_b, xb, tb, RR, CC, CC, stream);

  // ---- u = fkan1(LN2(t)) + t ; x_tr = x + u
  ln_kernel<<<RR, 128, 0, stream>>>(tb, ln2_g, ln2_b, lnf, nullptr);
  fkan_block(lnf, f1_ng, f1_nb, f1_bw0, f1_sw0, f1_sc0, f1_bw1, f1_sw1, f1_sc1,
             tb, ub, lnf2, acatb, hbuf, wcat0b, wcat1b, stream);
  add_kernel<<<cdiv(EL, 256), 256, 0, stream>>>(xb, ub, xtrb, (int)EL);

  // ---- m = mamba(LN3(x_tr)) + x_tr
  ln_kernel<<<RR, 128, 0, stream>>>(xtrb, ln3_g, ln3_b, nullptr, lnbf);
  cvt_bf16_kernel<<<cdiv(CC * CC, 256), 256, 0, stream>>>(in_proj_w, winpb, CC * CC);
  cvt_bf16_kernel<<<cdiv(CC * CC, 256), 256, 0, stream>>>(out_proj_w, woutpb, CC * CC);
  launch_gemm(lnbf, winpb, nullptr, nullptr, xzb, RR, CC, CC, stream);
  conv1d_kernel<<<cdiv((size_t)RR * DH, 256), 256, 0, stream>>>(xzb, conv_x_w, conv_z_w, xlb, zlb);
  xproj_kernel<<<cdiv((size_t)RR * 24, 256), 256, 0, stream>>>(xlb, x_proj_w, xdblb);
  dt_kernel<<<cdiv((size_t)RR * DH, 256), 256, 0, stream>>>(xdblb, dt_proj_w, dt_proj_b, deltab);
  aneg_kernel<<<2, 256, 0, stream>>>(A_log, anegb);
  scan1_kernel<<<(BB * DH * DS * NCH) / 256, 256, 0, stream>>>(deltab, xdblb, xlb, anegb, scanP, scanHc);
  scan2_kernel<<<(BB * DH * DS) / 256, 256, 0, stream>>>(scanP, scanHc, scanHi);
  scan3_kernel<<<(BB * DH * DS * NCH) / 256, 256, 0, stream>>>(deltab, xdblb, xlb, anegb, scanHi, yb);
  ycat_kernel<<<cdiv((size_t)RR * DH, 256), 256, 0, stream>>>(yb, xlb, zlb, D_param, ycatb);
  launch_gemm(ycatb, woutpb, nullptr, xtrb, mb, RR, CC, CC, stream);

  // ---- nm = fkan2(LN4(m)) + m ; out = x_tr + nm (transposed to [B,C,N])
  ln_kernel<<<RR, 128, 0, stream>>>(mb, ln4_g, ln4_b, lnf, nullptr);
  fkan_block(lnf, f2_ng, f2_nb, f2_bw0, f2_sw0, f2_sc0, f2_bw1, f2_sw1, f2_sc1,
             mb, nmb, lnf2, acatb, hbuf, wcat0b, wcat1b, stream);
  final_kernel<<<cdiv(EL, 256), 256, 0, stream>>>(xtrb, nmb, (float*)d_out);
}
